// MLA_59072980189845
// MI455X (gfx1250) — matrix operand order repaired and checked by reading, not compiled, not run on hardware
//
#include <hip/hip_runtime.h>
#include <hip/hip_bf16.h>
#include <math.h>
#include <stdint.h>

// ---------------------------------------------------------------------------
// MLA forward for MI455X (gfx1250, wave32, WMMA).
// Matmuls: v_wmma_f32_16x16x32_bf16 (f32 accum).
// GEMM staging: double-buffered global_load_async_to_lds_b128 (ASYNCcnt).
// ---------------------------------------------------------------------------

typedef __bf16 bf16;
typedef __bf16 bf16x16 __attribute__((ext_vector_type(16)));
typedef float  f32x8   __attribute__((ext_vector_type(8)));
typedef unsigned int u32x4 __attribute__((ext_vector_type(4)));
typedef int v4i __attribute__((ext_vector_type(4)));

#define H_   16
#define DN_  128
#define DR_  64
#define DV_  128
#define DK_  192     // DN+DR
#define R_   512
#define DIM_ 2048
#define S_   2048
#define B_   2
#define NTOK (B_*S_) // 4096
#define KVA_N 640    // 576 padded up to a multiple of 128
#define SCALE_ 0.07216878364870322f  // 1/sqrt(192)

#if defined(__HIP_DEVICE_COMPILE__) && \
    __has_builtin(__builtin_amdgcn_global_load_async_to_lds_b128) && \
    __has_builtin(__builtin_amdgcn_s_wait_asynccnt)
#define HAS_ASYNC 1
#else
#define HAS_ASYNC 0
#endif

#if HAS_ASYNC
typedef __attribute__((address_space(1))) v4i* as1_v4i_ptr;
typedef __attribute__((address_space(3))) v4i* as3_v4i_ptr;
#endif

union FragAB { bf16x16 v; u32x4 q[2]; };

static __device__ __forceinline__ f32x8 wmma_bf16(bf16x16 a, bf16x16 b, f32x8 c) {
  return __builtin_amdgcn_wmma_f32_16x16x32_bf16(false, a, false, b, (short)0, c,
                                                 false, false);
}

// 16-byte global -> LDS copy (async on gfx1250, sync fallback otherwise).
static __device__ __forceinline__ void async_copy16(const bf16* g, bf16* l) {
#if HAS_ASYNC
  __builtin_amdgcn_global_load_async_to_lds_b128(
      (as1_v4i_ptr)(uintptr_t)g,
      (as3_v4i_ptr)(unsigned int)(uintptr_t)l,
      0, 0);
#else
  *(u32x4*)l = *(const u32x4*)g;
#endif
}

// --------------------------- elementwise casts -----------------------------

__global__ void k_cast_bf16(const float* __restrict__ in, bf16* __restrict__ out, int n) {
  int i = blockIdx.x * blockDim.x + threadIdx.x;
  int stride = gridDim.x * blockDim.x;
  for (; i < n; i += stride) out[i] = (bf16)in[i];
}

// in: [K][N] f32 row-major -> out: [N][K] bf16 row-major (out may be wider than N)
__global__ void k_transpose_bf16(const float* __restrict__ in, bf16* __restrict__ out,
                                 int K, int N) {
  int i = blockIdx.x * blockDim.x + threadIdx.x;
  int total = K * N;
  int stride = gridDim.x * blockDim.x;
  for (; i < total; i += stride) {
    int k = i / N, n = i - k * N;
    out[(size_t)n * K + k] = (bf16)in[i];
  }
}

// ------------------------------- WMMA GEMM ---------------------------------
// C[M][N] f32 = A[M][K] bf16 @ BT[N][K]^T bf16.
// Block tile 128x128, 8 waves (4m x 2n), wave tile 32x64 (2x4 WMMA tiles).
// K-step 32, double-buffered async-to-LDS staging.

__global__ __launch_bounds__(256) void k_gemm_bf16(
    const bf16* __restrict__ A, const bf16* __restrict__ BT,
    float* __restrict__ C, int M, int N, int K) {
  __shared__ bf16 sA[2][128][32];
  __shared__ bf16 sB[2][128][32];
  const int tid  = threadIdx.x;
  const int wave = tid >> 5, lane = tid & 31;
  const int half = lane >> 4, r = lane & 15;
  const int wm = wave & 3, wn = wave >> 2;
  const int m0 = blockIdx.y * 128, n0 = blockIdx.x * 128;

  f32x8 acc[2][4];
  for (int mt = 0; mt < 2; ++mt)
    for (int nt = 0; nt < 4; ++nt)
      for (int e = 0; e < 8; ++e) acc[mt][nt][e] = 0.0f;

  const int row = tid >> 1;          // 0..127
  const int col = (tid & 1) * 16;    // 0 or 16
  const bf16* Ag = A  + (size_t)(m0 + row) * K + col;
  const bf16* Bg = BT + (size_t)(n0 + row) * K + col;

  auto stage = [&](int buf, int k0) {
    async_copy16(Ag + k0,     &sA[buf][row][col]);
    async_copy16(Ag + k0 + 8, &sA[buf][row][col + 8]);
    async_copy16(Bg + k0,     &sB[buf][row][col]);
    async_copy16(Bg + k0 + 8, &sB[buf][row][col + 8]);
  };

  stage(0, 0);
  for (int k0 = 0; k0 < K; k0 += 32) {
    const int  cur      = (k0 >> 5) & 1;
    const bool has_next = (k0 + 32) < K;
    if (has_next) stage(cur ^ 1, k0 + 32);      // prefetch next tile into other buf
#if HAS_ASYNC
    // Async loads retire in order: cnt<=4 means the current stage has landed.
    if (has_next) __builtin_amdgcn_s_wait_asynccnt(4);
    else          __builtin_amdgcn_s_wait_asynccnt(0);
#endif
    __syncthreads();

    FragAB fa[2], fb[4];
    for (int mt = 0; mt < 2; ++mt) {
      fa[mt].q[0] = *(const u32x4*)&sA[cur][wm * 32 + mt * 16 + r][half * 16];
      fa[mt].q[1] = *(const u32x4*)&sA[cur][wm * 32 + mt * 16 + r][half * 16 + 8];
    }
    for (int nt = 0; nt < 4; ++nt) {
      fb[nt].q[0] = *(const u32x4*)&sB[cur][wn * 64 + nt * 16 + r][half * 16];
      fb[nt].q[1] = *(const u32x4*)&sB[cur][wn * 64 + nt * 16 + r][half * 16 + 8];
    }
    for (int mt = 0; mt < 2; ++mt)
      for (int nt = 0; nt < 4; ++nt)
        acc[mt][nt] = wmma_bf16(fa[mt].v, fb[nt].v, acc[mt][nt]);
    __syncthreads();   // protect cur buffer before it is overwritten next round
  }

  for (int mt = 0; mt < 2; ++mt)
    for (int nt = 0; nt < 4; ++nt)
      for (int e = 0; e < 8; ++e) {
        int crow = m0 + wm * 32 + mt * 16 + e + half * 8;
        int ccol = n0 + wn * 64 + nt * 16 + r;
        C[(size_t)crow * N + ccol] = acc[mt][nt][e];
      }
}

// --------------------- RoPE + RMS-norm + q repack --------------------------
// qbuf [NTOK][3072] f32, kva [NTOK][KVA_N] f32 (640-wide, 576 valid) ->
//   qf  [B*H][S][192] bf16, kvn [NTOK][512] bf16, kpe [NTOK][64] bf16

__global__ __launch_bounds__(256) void k_rope_norm(
    const float* __restrict__ qbuf, const float* __restrict__ kva,
    const float* __restrict__ kv_w, bf16* __restrict__ qf,
    bf16* __restrict__ kvn, bf16* __restrict__ kpe) {
  __shared__ float red[256];
  __shared__ float s_rs;
  const int tok  = blockIdx.x;
  const int b    = tok / S_, spos = tok - b * S_;
  const int tid  = threadIdx.x;
  const float* kva_t = kva + (size_t)tok * KVA_N;

  float ss = 0.f;
  for (int i = tid; i < R_; i += 256) { float v = kva_t[i]; ss += v * v; }
  red[tid] = ss;
  __syncthreads();
  for (int s = 128; s > 0; s >>= 1) {
    if (tid < s) red[tid] += red[tid + s];
    __syncthreads();
  }
  if (tid == 0) s_rs = rsqrtf(red[0] / (float)R_ + 1e-6f);
  __syncthreads();
  const float rs = s_rs;

  for (int i = tid; i < R_; i += 256)
    kvn[(size_t)tok * R_ + i] = (bf16)(kva_t[i] * rs * kv_w[i]);

  if (tid < 32) {  // k_pe rope: 32 (even, odd) pairs
    float inv = __powf(10000.f, -(float)tid / 32.f);
    float ang = (float)spos * inv;
    float c = __cosf(ang), sn = __sinf(ang);
    float x0 = kva_t[R_ + 2 * tid], x1 = kva_t[R_ + 2 * tid + 1];
    kpe[(size_t)tok * DR_ + 2 * tid]     = (bf16)(x0 * c - x1 * sn);
    kpe[(size_t)tok * DR_ + 2 * tid + 1] = (bf16)(x0 * sn + x1 * c);
  }

  const float* q_t = qbuf + (size_t)tok * (H_ * DK_);
  for (int idx = tid; idx < H_ * DK_; idx += 256) {
    int h = idx / DK_, d = idx - h * DK_;
    float val;
    if (d < DN_) {
      val = q_t[idx];
    } else {
      int local = d - DN_, pair = local >> 1;
      float inv = __powf(10000.f, -(float)pair / 32.f);
      float ang = (float)spos * inv;
      float c = __cosf(ang), sn = __sinf(ang);
      float x0 = q_t[h * DK_ + DN_ + 2 * pair];
      float x1 = q_t[h * DK_ + DN_ + 2 * pair + 1];
      val = (local & 1) ? (x0 * sn + x1 * c) : (x0 * c - x1 * sn);
    }
    qf[(((size_t)(b * H_ + h)) * S_ + spos) * DK_ + d] = (bf16)val;
  }
}

// ---------------------- k/v repack (head-major, v transposed) --------------
// kvb [NTOK][H*256] f32 -> kf [B*H][S][192] bf16, vT [B*H][128][S] bf16

__global__ __launch_bounds__(128) void k_repack(
    const float* __restrict__ kvb, const bf16* __restrict__ kpe,
    bf16* __restrict__ kfb, bf16* __restrict__ vT) {
  const int tok = blockIdx.x, h = blockIdx.y, d = threadIdx.x;
  const int b = tok / S_, spos = tok - b * S_;
  const int bh = b * H_ + h;
  const float* src = kvb + (size_t)tok * (H_ * 256) + h * 256;
  kfb[((size_t)bh * S_ + spos) * DK_ + d] = (bf16)src[d];
  vT[((size_t)bh * DV_ + d) * S_ + spos]  = (bf16)src[DN_ + d];
  if (d < DR_)
    kfb[((size_t)bh * S_ + spos) * DK_ + DN_ + d] = kpe[(size_t)tok * DR_ + d];
}

// -------------------------- flash attention --------------------------------
// One wave per 16-row query tile; online softmax; WMMA for S=QK^T and P@V.
// grid (S/128, H, B), 256 threads (8 waves).

__global__ __launch_bounds__(256) void k_mla_attn(
    const bf16* __restrict__ qf, const bf16* __restrict__ kf,
    const bf16* __restrict__ vT, bf16* __restrict__ attn) {
  __shared__ bf16 pbuf[8][16][32];   // per-wave P tile (D-layout -> A-layout bounce)
  const int tid  = threadIdx.x;
  const int wave = tid >> 5, lane = tid & 31;
  const int half = lane >> 4, r = lane & 15;
  const int h = blockIdx.y, b = blockIdx.z;
  const int bh = b * H_ + h;
  const int m0 = blockIdx.x * 128 + wave * 16;

  // Q fragments (A-layout), K = 192 -> 6 fragments
  const bf16* qbase = qf + ((size_t)bh * S_ + m0 + r) * DK_;
  FragAB qfr[6];
  for (int i = 0; i < 6; ++i) {
    qfr[i].q[0] = *(const u32x4*)(qbase + i * 32 + half * 16);
    qfr[i].q[1] = *(const u32x4*)(qbase + i * 32 + half * 16 + 8);
  }

  f32x8 O[8];
  float mi[8], li[8];
  for (int n = 0; n < 8; ++n)
    for (int e = 0; e < 8; ++e) O[n][e] = 0.f;
  for (int e = 0; e < 8; ++e) { mi[e] = -3.0e38f; li[e] = 0.f; }

  const int t_hi = m0 + 16;                 // exclusive causal bound
  for (int t0 = 0; t0 < t_hi; t0 += 32) {   // 32 keys per P@V step
    for (int sub = 0; sub < 2; ++sub) {
      const int ts = t0 + sub * 16;
      if (ts < t_hi) {
        f32x8 sc;
        for (int e = 0; e < 8; ++e) sc[e] = 0.f;
        const bf16* kbase = kf + ((size_t)bh * S_ + ts + r) * DK_;
        for (int i = 0; i < 6; ++i) {        // S tile = sum of 6 K=32 WMMAs
          FragAB kfr;
          kfr.q[0] = *(const u32x4*)(kbase + i * 32 + half * 16);
          kfr.q[1] = *(const u32x4*)(kbase + i * 32 + half * 16 + 8);
          sc = wmma_bf16(qfr[i].v, kfr.v, sc);
        }
        const int colk = ts + r;             // this lane's key index
        for (int e = 0; e < 8; ++e) {
          const int rowq = m0 + e + half * 8;
          float sv = sc[e] * SCALE_;
          if (colk > rowq) sv = -3.0e38f;    // causal mask
          float mx = sv;                     // row max over 16 lanes
          for (int off = 1; off < 16; off <<= 1)
            mx = fmaxf(mx, __shfl_xor(mx, off, 32));
          const float mnew = fmaxf(mi[e], mx);
          const float corr = __expf(mi[e] - mnew);
          const float pe   = __expf(sv - mnew);
          float rsum = pe;                   // row sum over 16 lanes
          for (int off = 1; off < 16; off <<= 1)
            rsum += __shfl_xor(rsum, off, 32);
          li[e] = li[e] * corr + rsum;
          mi[e] = mnew;
          for (int n = 0; n < 8; ++n) O[n][e] *= corr;
          pbuf[wave][e + half * 8][sub * 16 + r] = (bf16)pe;
        }
      } else {
        for (int e = 0; e < 8; ++e)
          pbuf[wave][e + half * 8][sub * 16 + r] = (bf16)0.0f;
      }
    }
    // P A-fragment from LDS (same wave, DS ops are in-order)
    FragAB pf;
    pf.q[0] = *(const u32x4*)&pbuf[wave][r][half * 16];
    pf.q[1] = *(const u32x4*)&pbuf[wave][r][half * 16 + 8];
    const bf16* vbase = vT + ((size_t)bh * DV_) * S_;
    for (int n = 0; n < 8; ++n) {            // O[16x128] += P[16x32] @ V[32x128]
      FragAB vf;
      const bf16* vr = vbase + (size_t)(n * 16 + r) * S_ + t0 + half * 16;
      vf.q[0] = *(const u32x4*)vr;
      vf.q[1] = *(const u32x4*)(vr + 8);
      O[n] = wmma_bf16(pf.v, vf.v, O[n]);
    }
  }

  for (int n = 0; n < 8; ++n)
    for (int e = 0; e < 8; ++e) {
      const int rowq = m0 + e + half * 8;
      const float val = O[n][e] / li[e];
      attn[((size_t)b * S_ + rowq) * (H_ * DV_) + h * DV_ + n * 16 + r] = (bf16)val;
    }
}

// ------------------------------ orchestration ------------------------------

extern "C" void kernel_launch(void* const* d_in, const int* in_sizes, int n_in,
                              void* d_out, int out_size, void* d_ws, size_t ws_size,
                              hipStream_t stream) {
  const float* x     = (const float*)d_in[0];
  const float* wq    = (const float*)d_in[1];
  const float* wkv_a = (const float*)d_in[2];
  const float* kv_w  = (const float*)d_in[3];
  const float* wkv_b = (const float*)d_in[4];
  const float* wo    = (const float*)d_in[5];
  float* out = (float*)d_out;

  char* p = (char*)d_ws;
  auto alloc = [&](size_t bytes) {
    char* q = p;
    p += (bytes + 255) & ~(size_t)255;
    return (void*)q;
  };
  bf16*  xb    = (bf16*) alloc((size_t)NTOK * DIM_ * 2);
  bf16*  wqT   = (bf16*) alloc((size_t)3072 * DIM_ * 2);
  bf16*  wkvaT = (bf16*) alloc((size_t)KVA_N * DIM_ * 2);  // rows 576..639 = pad
  bf16*  wkvbT = (bf16*) alloc((size_t)4096 * R_   * 2);
  bf16*  woT   = (bf16*) alloc((size_t)DIM_ * DIM_ * 2);
  float* qbuf  = (float*)alloc((size_t)NTOK * 3072 * 4);
  float* kva   = (float*)alloc((size_t)NTOK * KVA_N * 4);  // cols 576..639 = pad
  bf16*  kvn   = (bf16*) alloc((size_t)NTOK * R_   * 2);
  bf16*  kpe   = (bf16*) alloc((size_t)NTOK * DR_  * 2);
  float* kvb   = (float*)alloc((size_t)NTOK * 4096 * 4);
  bf16*  qfb   = (bf16*) alloc((size_t)B_ * H_ * S_ * DK_ * 2);
  bf16*  kfb   = (bf16*) alloc((size_t)B_ * H_ * S_ * DK_ * 2);
  bf16*  vT    = (bf16*) alloc((size_t)B_ * H_ * DV_ * S_ * 2);
  bf16*  attn  = (bf16*) alloc((size_t)NTOK * DIM_ * 2);

  k_cast_bf16<<<2048, 256, 0, stream>>>(x, xb, NTOK * DIM_);
  k_transpose_bf16<<<2048, 256, 0, stream>>>(wq,    wqT,   DIM_, 3072);
  k_transpose_bf16<<<1024, 256, 0, stream>>>(wkv_a, wkvaT, DIM_, 576);
  k_transpose_bf16<<<1024, 256, 0, stream>>>(wkv_b, wkvbT, R_,   4096);
  k_transpose_bf16<<<2048, 256, 0, stream>>>(wo,    woT,   DIM_, DIM_);

  dim3 g1(3072 / 128, NTOK / 128);
  k_gemm_bf16<<<g1, 256, 0, stream>>>(xb, wqT, qbuf, NTOK, 3072, DIM_);
  dim3 g2(KVA_N / 128, NTOK / 128);
  k_gemm_bf16<<<g2, 256, 0, stream>>>(xb, wkvaT, kva, NTOK, KVA_N, DIM_);

  k_rope_norm<<<NTOK, 256, 0, stream>>>(qbuf, kva, kv_w, qfb, kvn, kpe);

  dim3 g3(4096 / 128, NTOK / 128);
  k_gemm_bf16<<<g3, 256, 0, stream>>>(kvn, wkvbT, kvb, NTOK, 4096, R_);

  dim3 g4(NTOK, H_);
  k_repack<<<g4, 128, 0, stream>>>(kvb, kpe, kfb, vT);

  dim3 g5(S_ / 128, H_, B_);
  k_mla_attn<<<g5, 256, 0, stream>>>(qfb, kfb, vT, attn);

  dim3 g6(DIM_ / 128, NTOK / 128);
  k_gemm_bf16<<<g6, 256, 0, stream>>>(attn, woT, out, NTOK, DIM_, DIM_);

  (void)in_sizes; (void)n_in; (void)out_size; (void)ws_size;
}
